// MultiHeadAttention_39977555591216
// MI455X (gfx1250) — matrix operand order repaired and checked by reading, not compiled, not run on hardware
//
#include <hip/hip_runtime.h>
#include <hip/hip_bf16.h>
#include <math.h>
#include <stdint.h>

typedef __attribute__((ext_vector_type(16))) __bf16 v16bf;
typedef __attribute__((ext_vector_type(8)))  __bf16 v8bf;
typedef __attribute__((ext_vector_type(8)))  float  v8f;

#define EMB   768
#define HEADS 12
#define DH    64
#define BATCH 4
#define SEQ   2048
#define MTOT  (BATCH*SEQ)

// Build a 16-element bf16 WMMA fragment from two 16B-aligned 8-element halves.
static __device__ __forceinline__ v16bf mkfrag(const __bf16* lo, const __bf16* hi) {
  v8bf a = *(const v8bf*)lo;
  v8bf b = *(const v8bf*)hi;
  v16bf r;
#pragma unroll
  for (int i = 0; i < 8; ++i) { r[i] = a[i]; r[i + 8] = b[i]; }
  return r;
}

// ---- CDNA5 async global->LDS copies (ASYNCcnt-tracked, bypass VGPRs) ----
// GV addressing: per-lane 64-bit global address + 32-bit LDS dest address;
// inst_offset advances BOTH addresses (CDNA5 ISA).
static __device__ __forceinline__ void async_b128_x2(uint32_t lds, const void* g) {
  asm volatile("global_load_async_to_lds_b128 %0, %1, off\n\t"
               "global_load_async_to_lds_b128 %0, %1, off offset:16"
               :: "v"(lds), "v"((uint64_t)(uintptr_t)g) : "memory");
}
static __device__ __forceinline__ void async_b128_x4(uint32_t lds, const void* g) {
  asm volatile("global_load_async_to_lds_b128 %0, %1, off\n\t"
               "global_load_async_to_lds_b128 %0, %1, off offset:16\n\t"
               "global_load_async_to_lds_b128 %0, %1, off offset:32\n\t"
               "global_load_async_to_lds_b128 %0, %1, off offset:48"
               :: "v"(lds), "v"((uint64_t)(uintptr_t)g) : "memory");
}
#define WAIT_ASYNC(n) asm volatile("s_wait_asynccnt " #n ::: "memory")

// LDS byte offset of a __shared__ object: generic-pointer low 32 bits are the
// workgroup-relative LDS address on amdgcn.
static __device__ __forceinline__ uint32_t lds_off(const void* p) {
  return (uint32_t)(uintptr_t)p;
}

// ---------------------------------------------------------------- cast fp32->bf16
__global__ void cast_bf16_kernel(const float* __restrict__ in, __bf16* __restrict__ out, int n) {
  int i = blockIdx.x * blockDim.x + threadIdx.x;
  int stride = gridDim.x * blockDim.x;
  for (; i < n; i += stride) out[i] = (__bf16)in[i];
}

// ---------------------------------------------------------------- tiled WMMA GEMM
// C[M,N] = A[M,K] @ W[N,K]^T + bias[N]
// mode 0: write fp32 C row-major to outF
// mode 1: QKV epilogue: col = h*192 + d*3 + c  ->  scatter bf16 into q/k/v [b,h,n,d]
// Double-buffered async global->LDS staging pipelined against WMMA.
#define BK  32
#define LDT 40   // padded LDS tile row stride (elements)

__global__ __launch_bounds__(256)
void gemm_bf16_kernel(const __bf16* __restrict__ A, const __bf16* __restrict__ W,
                      const float* __restrict__ bias, int M, int Nn, int K,
                      int mode, float* __restrict__ outF,
                      __bf16* __restrict__ oq, __bf16* __restrict__ okk,
                      __bf16* __restrict__ ov)
{
  __shared__ __bf16 As[2][128 * LDT];
  __shared__ __bf16 Ws[2][128 * LDT];

  const int tid  = threadIdx.x;
  const int wave = tid >> 5, lane = tid & 31;
  const int hl   = lane >> 4, ln  = lane & 15;
  const int wm   = wave & 1,  wn  = wave >> 1;          // 2x4 wave grid: 64x32 per wave
  const int row0 = blockIdx.y * 128, col0 = blockIdx.x * 128;

  v8f zero = {0.f,0.f,0.f,0.f,0.f,0.f,0.f,0.f};
  v8f acc[4][2];
#pragma unroll
  for (int i = 0; i < 4; ++i)
#pragma unroll
    for (int j = 0; j < 2; ++j) acc[i][j] = zero;

  const int ldRow = tid >> 1, ldSeg = tid & 1;          // 128 rows x 2 segs of 16 elems

  // 4 async b128 per lane per tile (2 for A, 2 for W)
  auto issue_tile = [&](int k0, int buf) {
    async_b128_x2(lds_off(&As[buf][ldRow * LDT + ldSeg * 16]),
                  A + (size_t)(row0 + ldRow) * K + k0 + ldSeg * 16);
    async_b128_x2(lds_off(&Ws[buf][ldRow * LDT + ldSeg * 16]),
                  W + (size_t)(col0 + ldRow) * K + k0 + ldSeg * 16);
  };

  const int nk = K / BK;
  issue_tile(0, 0);
  for (int kt = 0; kt < nk; ++kt) {
    const int cur = kt & 1;
    if (kt + 1 < nk) {
      issue_tile((kt + 1) * BK, cur ^ 1);   // overlap next tile's DMA with compute
      WAIT_ASYNC(0x4);                      // tile kt's 4 asyncs complete (in-order)
    } else {
      WAIT_ASYNC(0x0);
    }
    __syncthreads();

    const __bf16* Ac = &As[cur][0];
    const __bf16* Wc = &Ws[cur][0];
    v16bf bfr[2];
#pragma unroll
    for (int tn = 0; tn < 2; ++tn) {
      const __bf16* p = &Wc[(wn * 32 + tn * 16 + ln) * LDT + hl * 8];
      bfr[tn] = mkfrag(p, p + 16);
    }
#pragma unroll
    for (int tm = 0; tm < 4; ++tm) {
      const __bf16* p = &Ac[(wm * 64 + tm * 16 + ln) * LDT + hl * 8];
      v16bf afr = mkfrag(p, p + 16);
#pragma unroll
      for (int tn = 0; tn < 2; ++tn)
        acc[tm][tn] = __builtin_amdgcn_wmma_f32_16x16x32_bf16(
            false, afr, false, bfr[tn], (short)0, acc[tm][tn], false, false);
    }
    __syncthreads();   // all waves done with buf[cur] before it is refilled
  }

#pragma unroll
  for (int tm = 0; tm < 4; ++tm) {
#pragma unroll
    for (int tn = 0; tn < 2; ++tn) {
      int n = col0 + wn * 32 + tn * 16 + ln;
      float bv = bias[n];
#pragma unroll
      for (int v = 0; v < 8; ++v) {
        int m = row0 + wm * 64 + tm * 16 + v + 8 * hl;
        float val = acc[tm][tn][v] + bv;
        if (mode == 0) {
          outF[(size_t)m * Nn + n] = val;
        } else {
          int h = n / 192, r = n % 192, d = r / 3, ch = r % 3;
          int bb = m >> 11, nn2 = m & 2047;
          size_t dst = ((size_t)((bb * HEADS + h) * SEQ + nn2)) * DH + d;
          __bf16 bvv = (__bf16)val;
          if (ch == 0) oq[dst] = bvv; else if (ch == 1) okk[dst] = bvv; else ov[dst] = bvv;
        }
      }
    }
  }
}

// ---------------------------------------------------------------- flash attention
// One block per (b, h, 128-query tile). 8 waves; wave w owns query rows w*16..w*16+15.
// Streams 128-key chunks: S = Q K^T (WMMA), online softmax (raw logits; /sqrt(768)
// applied AFTER softmax per the reference), O += P V (WMMA). Output bf16 [b,n,(h d)].
// K chunk staged via async global->LDS (overlaps the scalar V-transpose staging).
#define KC  128
#define LKB 72    // kbuf [k][d] stride
#define LVB 136   // vtbuf [d][k] stride
#define LPB 136   // pbuf [q][k] stride

__global__ __launch_bounds__(256)
void attn_kernel(const __bf16* __restrict__ qs, const __bf16* __restrict__ ks,
                 const __bf16* __restrict__ vs, __bf16* __restrict__ ao)
{
  __shared__ __bf16 kbuf[KC * LKB];
  __shared__ __bf16 vtbuf[DH * LVB];
  __shared__ __bf16 pbuf[128 * LPB];

  const int tid  = threadIdx.x;
  const int wave = tid >> 5, lane = tid & 31;
  const int hl   = lane >> 4, ln  = lane & 15;
  const int q0   = blockIdx.x * 128;
  const int h    = blockIdx.y, b = blockIdx.z;
  const int wq0  = wave * 16;
  const size_t headoff = ((size_t)(b * HEADS + h)) * SEQ * DH;
  const __bf16* Q  = qs + headoff;
  const __bf16* Kp = ks + headoff;
  const __bf16* Vp = vs + headoff;

  // Persistent Q A-fragments: [16 q x 64 d] -> two 16x32 fragments.
  v16bf qa[2];
#pragma unroll
  for (int f = 0; f < 2; ++f) {
    const __bf16* p = Q + (size_t)(q0 + wq0 + ln) * DH + f * 32 + hl * 8;
    qa[f] = mkfrag(p, p + 16);
  }

  v8f zero = {0.f,0.f,0.f,0.f,0.f,0.f,0.f,0.f};
  v8f oacc[4];
#pragma unroll
  for (int i = 0; i < 4; ++i) oacc[i] = zero;
  float mrun[8], lrun[8];
#pragma unroll
  for (int v = 0; v < 8; ++v) { mrun[v] = -__builtin_inff(); lrun[v] = 0.f; }

  const int cr = tid >> 1, cs = tid & 1;                 // K staging: (row, 32-elem seg)
  const int vk = tid & 127, vd0 = (tid >> 7) * 32;       // V transpose staging

  for (int c = 0; c < SEQ / KC; ++c) {
    const int kc0 = c * KC;
    __syncthreads();                       // protect LDS reuse vs previous P@V reads
    // stage K chunk via async DMA: kbuf[k][d], 64B contiguous on both sides
    async_b128_x4(lds_off(&kbuf[cr * LKB + cs * 32]),
                  Kp + (size_t)(kc0 + cr) * DH + cs * 32);
    {  // stage V chunk transposed: vtbuf[d][k]  (overlaps the async K DMA)
      const __bf16* src = Vp + (size_t)(kc0 + vk) * DH + vd0;
#pragma unroll
      for (int j = 0; j < 32; ++j) vtbuf[(vd0 + j) * LVB + vk] = src[j];
    }
    if (c + 1 < SEQ / KC)   // prefetch next V chunk (global_prefetch_b8)
      __builtin_prefetch(Vp + (size_t)(kc0 + KC + vk) * DH + vd0, 0, 0);
    WAIT_ASYNC(0x0);
    __syncthreads();

    // S strip [16 q x 128 k]: 8 C-tiles, 2 WMMA each (d = 0..31, 32..63)
    v8f sacc[8];
#pragma unroll
    for (int t = 0; t < 8; ++t) sacc[t] = zero;
#pragma unroll
    for (int f = 0; f < 2; ++f) {
#pragma unroll
      for (int t = 0; t < 8; ++t) {
        const __bf16* p = &kbuf[(t * 16 + ln) * LKB + f * 32 + hl * 8];
        v16bf kb = mkfrag(p, p + 16);
        sacc[t] = __builtin_amdgcn_wmma_f32_16x16x32_bf16(
            false, qa[f], false, kb, (short)0, sacc[t], false, false);
      }
    }

    // Online softmax per row (row v+8*hl lives in this lane's element v,
    // spread over the 16 lanes of this half -> shfl_xor 1/2/4/8 reduction).
#pragma unroll
    for (int v = 0; v < 8; ++v) {
      float mx = sacc[0][v];
#pragma unroll
      for (int t = 1; t < 8; ++t) mx = fmaxf(mx, sacc[t][v]);
#pragma unroll
      for (int m = 1; m < 16; m <<= 1) mx = fmaxf(mx, __shfl_xor(mx, m, 32));
      float mnew  = fmaxf(mrun[v], mx);
      float alpha = __expf(mrun[v] - mnew);
      mrun[v] = mnew;
#pragma unroll
      for (int dt = 0; dt < 4; ++dt) oacc[dt][v] *= alpha;
      float ps = 0.f;
      const int prow = (wq0 + v + 8 * hl) * LPB;
#pragma unroll
      for (int t = 0; t < 8; ++t) {
        float p = __expf(sacc[t][v] - mnew);
        ps += p;
        pbuf[prow + t * 16 + ln] = (__bf16)p;
      }
#pragma unroll
      for (int m = 1; m < 16; m <<= 1) ps += __shfl_xor(ps, m, 32);
      lrun[v] = lrun[v] * alpha + ps;
    }
    __syncthreads();                       // P C-layout -> A-layout via LDS

    // O strip [16 q x 64 d] += P[16 x 128] @ V[128 x 64]
#pragma unroll
    for (int ko = 0; ko < 4; ++ko) {
      const __bf16* p = &pbuf[(wq0 + ln) * LPB + ko * 32 + hl * 8];
      v16bf pa = mkfrag(p, p + 16);
#pragma unroll
      for (int dt = 0; dt < 4; ++dt) {
        const __bf16* p2 = &vtbuf[(dt * 16 + ln) * LVB + ko * 32 + hl * 8];
        v16bf vb = mkfrag(p2, p2 + 16);
        oacc[dt] = __builtin_amdgcn_wmma_f32_16x16x32_bf16(
            false, pa, false, vb, (short)0, oacc[dt], false, false);
      }
    }
  }

  // Final normalization: softmax / sqrt(EMB) applied after softmax (reference).
  const float scale = 27.712812921102035f;  // sqrt(768)
#pragma unroll
  for (int v = 0; v < 8; ++v) {
    int q = q0 + wq0 + v + 8 * hl;
    float inv = 1.f / (lrun[v] * scale);
    size_t base = ((size_t)(b * SEQ + q)) * EMB + h * DH;
#pragma unroll
    for (int dt = 0; dt < 4; ++dt)
      ao[base + dt * 16 + ln] = (__bf16)(oacc[dt][v] * inv);
  }
}

// ---------------------------------------------------------------- launch
extern "C" void kernel_launch(void* const* d_in, const int* in_sizes, int n_in,
                              void* d_out, int out_size, void* d_ws, size_t ws_size,
                              hipStream_t stream) {
  (void)in_sizes; (void)n_in; (void)out_size; (void)ws_size;
  const float* x      = (const float*)d_in[0];
  const float* qkv_w  = (const float*)d_in[1];
  const float* qkv_b  = (const float*)d_in[2];
  const float* proj_w = (const float*)d_in[3];
  const float* proj_b = (const float*)d_in[4];
  float* out = (float*)d_out;

  char* w = (char*)d_ws;
  size_t off = 0;
  auto carve = [&](size_t bytes) -> void* {
    void* p = w + off;
    off += (bytes + 255) & ~(size_t)255;
    return p;
  };
  __bf16* xh  = (__bf16*)carve((size_t)MTOT * EMB * 2);
  __bf16* qwh = (__bf16*)carve((size_t)3 * EMB * EMB * 2);
  __bf16* pwh = (__bf16*)carve((size_t)EMB * EMB * 2);
  __bf16* qsb = (__bf16*)carve((size_t)MTOT * EMB * 2);
  __bf16* ksb = (__bf16*)carve((size_t)MTOT * EMB * 2);
  __bf16* vsb = (__bf16*)carve((size_t)MTOT * EMB * 2);
  __bf16* aob = (__bf16*)carve((size_t)MTOT * EMB * 2);

  cast_bf16_kernel<<<4096, 256, 0, stream>>>(x, xh, MTOT * EMB);
  cast_bf16_kernel<<<2048, 256, 0, stream>>>(qkv_w, qwh, 3 * EMB * EMB);
  cast_bf16_kernel<<<1024, 256, 0, stream>>>(proj_w, pwh, EMB * EMB);

  // QKV GEMM: [8192 x 768] @ [2304 x 768]^T + bias, scatter to per-head Q/K/V
  gemm_bf16_kernel<<<dim3(3 * EMB / 128, MTOT / 128), 256, 0, stream>>>(
      xh, qwh, qkv_b, MTOT, 3 * EMB, EMB, 1, nullptr, qsb, ksb, vsb);

  // Attention: grid (query tiles, heads, batch)
  attn_kernel<<<dim3(SEQ / 128, HEADS, BATCH), 256, 0, stream>>>(qsb, ksb, vsb, aob);

  // Output projection: [8192 x 768] @ [768 x 768]^T + bias -> fp32 d_out
  gemm_bf16_kernel<<<dim3(EMB / 128, MTOT / 128), 256, 0, stream>>>(
      aob, pwh, proj_b, MTOT, EMB, EMB, 0, out, nullptr, nullptr, nullptr);
}
